// RGCN_79474074845320
// MI455X (gfx1250) — matrix operand order repaired and checked by reading, not compiled, not run on hardware
//
#include <hip/hip_runtime.h>
#include <hip/hip_bf16.h>

typedef __attribute__((ext_vector_type(16))) __bf16 v16bf;
typedef __attribute__((ext_vector_type(8)))  float  v8f;
typedef __attribute__((ext_vector_type(4)))  float  f32x4;
typedef __attribute__((ext_vector_type(4)))  int    i32x4;
typedef __attribute__((ext_vector_type(4)))  unsigned int u32x4;

#define O_TOT 2048
#define M_TOT 512
#define FIN   64
#define FMAS  64
#define DOUT  32
#define HDIM  128

// padded LDS strides (all chosen so row stride (bytes) % 16 == 0 and
// stride/4 % 32 == 4 banks -> 16 lanes * 4-bank footprint = conflict-free)
#define HMAS_S 520   // 512 + 8
#define W1T_S  72    // 64 + 8
#define W2T_S  136   // 128 + 8
#define WEFF_S 72
#define STG1_S 136   // stage stride for [16][128] hidden tile
#define STG3_S 72    // stage stride for [16][64] agg tile

#define LDS_HMAST 0
#define LDS_W1T   (LDS_HMAST + FMAS * HMAS_S * 2)          // 66560
#define LDS_W2T   (LDS_W1T   + HDIM * W1T_S * 2)           // +18432
#define LDS_WEFFT (LDS_W2T   + DOUT * W2T_S * 2)           // +8704
#define LDS_OMW2  (LDS_WEFFT + HDIM * WEFF_S * 2)          // +18432
#define LDS_STAGE (LDS_OMW2  + HDIM * 4)                   // +512
#define LDS_TOTAL (LDS_STAGE + 8 * 16 * STG1_S * 2)        // 147456 = 144 KB

__device__ __forceinline__ float elu_f(float x) { return x > 0.f ? x : (__expf(x) - 1.f); }

// 16 contiguous bf16 (32B, 16B-aligned) -> fragment: two ds_load_b128
__device__ __forceinline__ v16bf ld_bfrag(const __bf16* p) {
    u32x4 lo = *(const u32x4*)(p);
    u32x4 hi = *(const u32x4*)(p + 8);
    v16bf r;
    __builtin_memcpy(&r, &lo, 16);
    __builtin_memcpy(((char*)&r) + 16, &hi, 16);
    return r;
}
// A fragment: 8 bf16 at p (K=base..base+7), 8 bf16 at p+16 (K=base+16..base+23)
__device__ __forceinline__ v16bf ld_afrag(const __bf16* p) {
    u32x4 lo = *(const u32x4*)(p);
    u32x4 hi = *(const u32x4*)(p + 16);
    v16bf r;
    __builtin_memcpy(&r, &lo, 16);
    __builtin_memcpy(((char*)&r) + 16, &hi, 16);
    return r;
}

// ---------------------------------------------------------------------------
// Prep: fold fc_edge into (k,c) scalars; fold (coeff@v) with om_w1 into
// W_eff[64][128] (bf16).
// ---------------------------------------------------------------------------
__global__ void rgcn_prep(const float* __restrict__ v, const float* __restrict__ coeff,
                          const float* __restrict__ e_w1, const float* __restrict__ e_b1,
                          const float* __restrict__ e_w2, const float* __restrict__ e_b2,
                          const float* __restrict__ om_w1,
                          float* __restrict__ ws_scal, unsigned short* __restrict__ ws_weff) {
    int t = threadIdx.x;
    if (t == 0) {
        float k = 0.f, c = 0.f;
        for (int h = 0; h < HDIM; ++h) { k += e_w1[h] * e_w2[h]; c += e_b1[h] * e_w2[h]; }
        ws_scal[0] = k;
        ws_scal[1] = c + e_b2[0];
    }
    for (int idx = t; idx < FMAS * HDIM; idx += blockDim.x) {
        int f = idx >> 7, h = idx & 127;
        float acc = 0.f;
        for (int t3 = 0; t3 < 3; ++t3) {
            float c0 = coeff[t3 * 3 + 0], c1 = coeff[t3 * 3 + 1], c2 = coeff[t3 * 3 + 2];
            for (int e = 0; e < DOUT; ++e) {
                float wr = c0 * v[(0 * FMAS + f) * DOUT + e]
                         + c1 * v[(1 * FMAS + f) * DOUT + e]
                         + c2 * v[(2 * FMAS + f) * DOUT + e];
                acc = fmaf(wr, om_w1[(t3 * DOUT + e) * HDIM + h], acc);
            }
        }
        __bf16 bv = (__bf16)acc;
        unsigned short u;
        __builtin_memcpy(&u, &bv, 2);
        ws_weff[idx] = u;
    }
}

// ---------------------------------------------------------------------------
// Main fused kernel. 256 threads = 8 waves; each wave owns 16 rows of O.
// ---------------------------------------------------------------------------
__global__ __launch_bounds__(256) void rgcn_main(
    const float* __restrict__ feat0, const float* __restrict__ feat1,
    const float* __restrict__ feat2, const int* __restrict__ adj,
    const long long* __restrict__ bidx,
    const float* __restrict__ w0_w1g, const float* __restrict__ w0_b1,
    const float* __restrict__ w0_w2g, const float* __restrict__ w0_b2,
    const float* __restrict__ om_b1,  const float* __restrict__ om_w2g,
    const float* __restrict__ om_b2,
    const float* __restrict__ ws_scal, const unsigned short* __restrict__ ws_weff,
    float* __restrict__ out) {
    extern __shared__ char smem[];
    __bf16* s_hmasT = (__bf16*)(smem + LDS_HMAST);  // [64][520]  (f-major, K-contig)
    __bf16* s_w1T   = (__bf16*)(smem + LDS_W1T);    // [128][72]
    __bf16* s_w2T   = (__bf16*)(smem + LDS_W2T);    // [32][136]
    __bf16* s_weffT = (__bf16*)(smem + LDS_WEFFT);  // [128][72]
    float*  s_omw2  = (float*)(smem + LDS_OMW2);    // [128]
    __bf16* s_stage = (__bf16*)(smem + LDS_STAGE);  // [8 waves][16][STG1_S]

    const int tid  = threadIdx.x;
    const int lane = tid & 31;
    const int wv   = tid >> 5;
    const int hi   = lane >> 4;   // lane half per CDNA5 WMMA layout
    const int ln   = lane & 15;

    const long long bi = bidx[blockIdx.y];
    const float* f1b  = feat1 + bi * (long long)(M_TOT * FMAS);
    const float* f0b  = feat0 + bi * (long long)(O_TOT * FIN);
    const float* f2b  = feat2 + bi * (long long)(O_TOT * M_TOT);
    const int*   adjb = adj   + bi * (long long)(O_TOT * M_TOT);

    // ---- stage operands into LDS, transposed so K is contiguous ----
    for (int i = tid; i < M_TOT * FMAS; i += 256) {         // h_mas[m][f] -> [f][m]
        int m = i >> 6, f = i & 63;
        s_hmasT[f * HMAS_S + m] = (__bf16)f1b[i];
    }
    for (int i = tid; i < FIN * HDIM; i += 256) {           // W1[k][h] -> [h][k]
        int k = i >> 7, h = i & 127;
        s_w1T[h * W1T_S + k] = (__bf16)w0_w1g[i];
    }
    for (int i = tid; i < HDIM * DOUT; i += 256) {          // W2[k][n] -> [n][k]
        int k = i >> 5, n = i & 31;
        s_w2T[n * W2T_S + k] = (__bf16)w0_w2g[i];
    }
    for (int i = tid; i < FMAS * HDIM; i += 256) {          // Weff[f][h] -> [h][f]
        int f = i >> 7, h = i & 127;
        unsigned short u = ws_weff[i];
        __bf16 bv;
        __builtin_memcpy(&bv, &u, 2);
        s_weffT[h * WEFF_S + f] = bv;
    }
    if (tid < HDIM) s_omw2[tid] = om_w2g[tid];
    __syncthreads();

    const float k_e = ws_scal[0];
    const float c_e = ws_scal[1];

    const int o_base = blockIdx.x * 128 + wv * 16;
    const int rowA   = ln;
    const int oA     = o_base + rowA;
    __bf16* stg = s_stage + wv * 16 * STG1_S;

    // =====================================================================
    // Path 1: feat_opt = ELU(h_opt @ W1 + b1) @ W2 + b2
    // =====================================================================
    v16bf aop[2];
#pragma unroll
    for (int c = 0; c < 2; ++c) {
        const f32x4* s0 = (const f32x4*)(f0b + (long long)oA * FIN + c * 32 + hi * 8);
        f32x4 x0 = s0[0], x1 = s0[1];   // K = base .. base+7
        f32x4 x2 = s0[4], x3 = s0[5];   // K = base+16 .. base+23
#pragma unroll
        for (int j = 0; j < 4; ++j) {
            aop[c][j]      = (__bf16)x0[j];
            aop[c][4 + j]  = (__bf16)x1[j];
            aop[c][8 + j]  = (__bf16)x2[j];
            aop[c][12 + j] = (__bf16)x3[j];
        }
    }

#pragma unroll
    for (int nt = 0; nt < 8; ++nt) {
        v8f h = {0.f, 0.f, 0.f, 0.f, 0.f, 0.f, 0.f, 0.f};
#pragma unroll
        for (int c = 0; c < 2; ++c) {
            v16bf bfr = ld_afrag(s_w1T + (nt * 16 + ln) * W1T_S + c * 32 + hi * 8);
            h = __builtin_amdgcn_wmma_f32_16x16x32_bf16(false, aop[c], false, bfr,
                                                        (short)0, h, false, false);
        }
        const float b1v = w0_b1[nt * 16 + ln];
#pragma unroll
        for (int r = 0; r < 8; ++r) {
            float x = elu_f(h[r] + b1v);
            stg[(r + 8 * hi) * STG1_S + nt * 16 + ln] = (__bf16)x;
        }
    }
    asm volatile("s_wait_dscnt 0" ::: "memory");

    v8f fo[2];
#pragma unroll
    for (int nt = 0; nt < 2; ++nt) fo[nt] = v8f{0.f, 0.f, 0.f, 0.f, 0.f, 0.f, 0.f, 0.f};
#pragma unroll
    for (int c = 0; c < 4; ++c) {
        v16bf a2 = ld_afrag(stg + rowA * STG1_S + c * 32 + hi * 8);
#pragma unroll
        for (int nt = 0; nt < 2; ++nt) {
            v16bf bfr = ld_afrag(s_w2T + (nt * 16 + ln) * W2T_S + c * 32 + hi * 8);
            fo[nt] = __builtin_amdgcn_wmma_f32_16x16x32_bf16(false, a2, false, bfr,
                                                             (short)0, fo[nt], false, false);
        }
    }
#pragma unroll
    for (int nt = 0; nt < 2; ++nt) {
        const float b2v = w0_b2[nt * 16 + ln];
#pragma unroll
        for (int r = 0; r < 8; ++r) fo[nt][r] += b2v;
    }

    // =====================================================================
    // Path 2 (hot): agg_in = ((feat2*k + c) * mask) @ h_mas
    // Single non-temporal pass over the 268 MB feat2/adj stream.
    // =====================================================================
    v8f agg[4];
#pragma unroll
    for (int ft = 0; ft < 4; ++ft) agg[ft] = v8f{0.f, 0.f, 0.f, 0.f, 0.f, 0.f, 0.f, 0.f};

    const float* f2row = f2b  + (long long)oA * M_TOT;
    const int*   adrow = adjb + (long long)oA * M_TOT;
    for (int m0 = 0; m0 < M_TOT; m0 += 32) {
        const f32x4* fp = (const f32x4*)(f2row + m0 + hi * 8);
        const i32x4* ap = (const i32x4*)(adrow + m0 + hi * 8);
        f32x4 fx0 = __builtin_nontemporal_load(fp);
        f32x4 fx1 = __builtin_nontemporal_load(fp + 1);
        f32x4 fx2 = __builtin_nontemporal_load(fp + 4);
        f32x4 fx3 = __builtin_nontemporal_load(fp + 5);
        i32x4 ax0 = __builtin_nontemporal_load(ap);
        i32x4 ax1 = __builtin_nontemporal_load(ap + 1);
        i32x4 ax2 = __builtin_nontemporal_load(ap + 4);
        i32x4 ax3 = __builtin_nontemporal_load(ap + 5);
        v16bf aw;
#pragma unroll
        for (int j = 0; j < 4; ++j) {
            aw[j]      = (__bf16)(ax0[j] == 1 ? fmaf(fx0[j], k_e, c_e) : 0.f);
            aw[4 + j]  = (__bf16)(ax1[j] == 1 ? fmaf(fx1[j], k_e, c_e) : 0.f);
            aw[8 + j]  = (__bf16)(ax2[j] == 1 ? fmaf(fx2[j], k_e, c_e) : 0.f);
            aw[12 + j] = (__bf16)(ax3[j] == 1 ? fmaf(fx3[j], k_e, c_e) : 0.f);
        }
#pragma unroll
        for (int ft = 0; ft < 4; ++ft) {
            v16bf bfr = ld_afrag(s_hmasT + (ft * 16 + ln) * HMAS_S + m0 + hi * 8);
            agg[ft] = __builtin_amdgcn_wmma_f32_16x16x32_bf16(false, aw, false, bfr,
                                                              (short)0, agg[ft], false, false);
        }
    }

    // =====================================================================
    // Path 3: mas_w = ELU(agg_in @ W_eff + om_b1) @ om_w2 + om_b2
    // =====================================================================
#pragma unroll
    for (int ft = 0; ft < 4; ++ft) {
#pragma unroll
        for (int r = 0; r < 8; ++r)
            stg[(r + 8 * hi) * STG3_S + ft * 16 + ln] = (__bf16)agg[ft][r];
    }
    asm volatile("s_wait_dscnt 0" ::: "memory");

    float part[8];
#pragma unroll
    for (int r = 0; r < 8; ++r) part[r] = 0.f;

#pragma unroll
    for (int nt = 0; nt < 8; ++nt) {
        v8f h2 = {0.f, 0.f, 0.f, 0.f, 0.f, 0.f, 0.f, 0.f};
#pragma unroll
        for (int c = 0; c < 2; ++c) {
            v16bf a3  = ld_afrag(stg + rowA * STG3_S + c * 32 + hi * 8);
            v16bf bfr = ld_afrag(s_weffT + (nt * 16 + ln) * WEFF_S + c * 32 + hi * 8);
            h2 = __builtin_amdgcn_wmma_f32_16x16x32_bf16(false, a3, false, bfr,
                                                         (short)0, h2, false, false);
        }
        const float ob1 = om_b1[nt * 16 + ln];
        const float wv2 = s_omw2[nt * 16 + ln];
#pragma unroll
        for (int r = 0; r < 8; ++r) part[r] = fmaf(elu_f(h2[r] + ob1), wv2, part[r]);
    }
#pragma unroll
    for (int s = 1; s < 16; s <<= 1) {
#pragma unroll
        for (int r = 0; r < 8; ++r) part[r] += __shfl_xor(part[r], s, 32);
    }
    const float ob2 = om_b2[0];

    // =====================================================================
    // out = leaky_relu(feat_opt + mas_w, 0.2) -- non-temporal stores
    // =====================================================================
    const long long obase = ((long long)blockIdx.y * O_TOT + o_base) * DOUT;
#pragma unroll
    for (int nt = 0; nt < 2; ++nt) {
#pragma unroll
        for (int r = 0; r < 8; ++r) {
            const int row = r + 8 * hi;
            const int col = nt * 16 + ln;
            float x = fo[nt][r] + part[r] + ob2;
            float y = x > 0.f ? x : 0.2f * x;
            __builtin_nontemporal_store(y, out + obase + (long long)row * DOUT + col);
        }
    }
}

// ---------------------------------------------------------------------------
extern "C" void kernel_launch(void* const* d_in, const int* in_sizes, int n_in,
                              void* d_out, int out_size, void* d_ws, size_t ws_size,
                              hipStream_t stream) {
    const float*     feat0 = (const float*)d_in[0];
    const float*     feat1 = (const float*)d_in[1];
    const float*     feat2 = (const float*)d_in[2];
    const int*       adj   = (const int*)d_in[3];
    const long long* bidx  = (const long long*)d_in[4];
    const float*     w0_w1 = (const float*)d_in[5];
    const float*     w0_b1 = (const float*)d_in[6];
    const float*     w0_w2 = (const float*)d_in[7];
    const float*     w0_b2 = (const float*)d_in[8];
    const float*     v     = (const float*)d_in[9];
    const float*     coeff = (const float*)d_in[10];
    const float*     e_w1  = (const float*)d_in[11];
    const float*     e_b1  = (const float*)d_in[12];
    const float*     e_w2  = (const float*)d_in[13];
    const float*     e_b2  = (const float*)d_in[14];
    const float*     om_w1 = (const float*)d_in[15];
    const float*     om_b1 = (const float*)d_in[16];
    const float*     om_w2 = (const float*)d_in[17];
    const float*     om_b2 = (const float*)d_in[18];
    float*           out   = (float*)d_out;

    float*          ws_scal = (float*)d_ws;
    unsigned short* ws_weff = (unsigned short*)((char*)d_ws + 16);

    const int b = in_sizes[4];

    rgcn_prep<<<1, 256, 0, stream>>>(v, coeff, e_w1, e_b1, e_w2, e_b2, om_w1,
                                     ws_scal, ws_weff);

    (void)hipFuncSetAttribute((const void*)rgcn_main,
                              hipFuncAttributeMaxDynamicSharedMemorySize,
                              (int)LDS_TOTAL);
    dim3 grid(O_TOT / 128, b);
    rgcn_main<<<grid, 256, LDS_TOTAL, stream>>>(feat0, feat1, feat2, adj, bidx,
                                                w0_w1, w0_b1, w0_w2, w0_b2,
                                                om_b1, om_w2, om_b2,
                                                ws_scal, ws_weff, out);
}
